// HORNN_76020921140340
// MI455X (gfx1250) — matrix operand order repaired and checked by reading, not compiled, not run on hardware
//
#include <hip/hip_runtime.h>
#include <hip/hip_bf16.h>

// ---------------- problem constants ----------------
#define HIDN   256
#define INN    64
#define OUTN   32
#define BB     64
#define TT     256
#define KTOT   65792            // HID*HID + HID
#define KTILES 2056             // KTOT / 32
#define NOISE_STD 0.05f
#define TAU       0.2f

typedef __attribute__((ext_vector_type(16))) __bf16 v16bf;
typedef __attribute__((ext_vector_type(8)))  float  v8f;

// ---------------- workspace layout (bytes) ----------------
#define WS_INPUTI 0             // [T][B][HID] f32                    (16,777,216)
#define WS_WC     16777216      // 2056 kt x 16 nt x 512 bf16         (33,685,504)
#define WS_AFRAG  50462720      // 4 mblk x 2056 kt x 512 bf16        ( 8,421,376)
#define WS_RBF    58884096      // [B][HID] bf16                      (    32,768)
#define WS_PART   58916864      // [64 ks][B][HID] f32                ( 4,194,304)

// ======================================================================
// One-time: input currents  inputI[t][b][h] = u[b,t,:] @ w_in_w[h,:] + b[h]
// ======================================================================
__global__ __launch_bounds__(256) void input_gemm_kernel(
    const float* __restrict__ u, const float* __restrict__ w_in_w,
    const float* __restrict__ w_in_b, float* __restrict__ inputI) {
  int tid = blockIdx.x * 256 + threadIdx.x;          // T*B*HID threads
  int h = tid & 255;
  int b = (tid >> 8) & 63;
  int t = tid >> 14;
  const float* urow = u + ((size_t)b * TT + t) * INN;
  const float* wrow = w_in_w + (size_t)h * INN;
  float s = w_in_b[h];
#pragma unroll 8
  for (int i = 0; i < INN; ++i) s += urow[i] * wrow[i];
  inputI[tid] = s;
}

// ======================================================================
// One-time: combined weights -> bf16 in WMMA B-fragment order.
// B frag (32x16 bf16), contiguous k order used here (same as A): lane n = lane&15, K = (lane>>4)*16 + e
// ======================================================================
__global__ __launch_bounds__(256) void convert_weights_kernel(
    const float* __restrict__ w3, const float* __restrict__ w2,
    __bf16* __restrict__ Wc) {
  int tid = blockIdx.x * 256 + threadIdx.x;          // KTILES*16*32 threads
  int lane = tid & 31;
  int nt   = (tid >> 5) & 15;
  int kt   = tid >> 9;
  if (kt >= KTILES) return;
  int n   = nt * 16 + (lane & 15);
  int kk0 = kt * 32 + (lane >> 4) * 16;
  alignas(16) __bf16 o[16];
  if (kk0 < 65536) {
    const float* src = w3 + (size_t)n * 65536 + kk0;
#pragma unroll
    for (int e = 0; e < 16; ++e) o[e] = (__bf16)src[e];
  } else {
    const float* src = w2 + (size_t)n * 256 + (kk0 - 65536);
#pragma unroll
    for (int e = 0; e < 16; ++e) o[e] = (__bf16)src[e];
  }
  __bf16* dst = Wc + ((size_t)(kt * 16 + nt)) * 512 + (size_t)lane * 16;
  *(uint4*)dst       = *(const uint4*)&o[0];
  *(uint4*)(dst + 8) = *(const uint4*)&o[8];
}

// ======================================================================
// Per step: state update (deterministic fixed-order split-K reduce) +
// r = tanh(x) in bf16.  t runs 0..T.
// ======================================================================
__global__ __launch_bounds__(256) void update_kernel(
    int t, const float* __restrict__ x0, const float* __restrict__ noise,
    const float* __restrict__ inputI, const float* __restrict__ part,
    float* __restrict__ traj, float* __restrict__ xfinal,
    __bf16* __restrict__ rbf) {
  int tid = blockIdx.x * 256 + threadIdx.x;          // B*HID = 16384
  int b = tid >> 8, h = tid & 255;
  float xcur;
  if (t == 0) {
    xcur = x0[tid];
  } else {
    float xold = (t == 1) ? x0[tid] : traj[((size_t)b * TT + (t - 2)) * HIDN + h];
    float rc = 0.f;
#pragma unroll 8
    for (int s = 0; s < 64; ++s) rc += part[(size_t)s * (BB * HIDN) + tid];
    float ii = inputI[(size_t)(t - 1) * (BB * HIDN) + tid];
    float nn = noise [(size_t)(t - 1) * (BB * HIDN) + tid];
    xcur = xold + NOISE_STD * nn + TAU * (-xold + rc + ii);
    traj[((size_t)b * TT + (t - 1)) * HIDN + h] = xcur;
    if (t == TT) xfinal[tid] = xcur;
  }
  if (t < TT) rbf[tid] = (__bf16)tanhf(xcur);
}

// ======================================================================
// Per step: build A-fragments of q' = [r (x) r , r] in WMMA A layout.
// A bf16 16x32 frag, contiguous k order used here (same as Wc): lane m = lane&15; K = {0..15} (lane<16) or
// {16..31} (lane>=16).
// ======================================================================
__global__ __launch_bounds__(256) void afrag_kernel(
    const __bf16* __restrict__ rbf, __bf16* __restrict__ Afrag) {
  int tid  = blockIdx.x * 256 + threadIdx.x;         // 4*KTILES*64 threads
  int half = tid & 1;
  int lane = (tid >> 1) & 31;
  int rest = tid >> 6;                               // mblk*KTILES + kt
  int kt   = rest % KTILES;
  int mblk = rest / KTILES;
  int b    = mblk * 16 + (lane & 15);
  int Kbase = (lane < 16) ? (half ? 8 : 0) : (half ? 24 : 16);
  int kk0   = kt * 32 + Kbase;
  const __bf16* rrow = rbf + (size_t)b * 256;
  alignas(16) __bf16 o[8];
  if (kk0 < 65536) {
    float ri = (float)rrow[kk0 >> 8];
    int   k0 = kk0 & 255;
#pragma unroll
    for (int j = 0; j < 8; ++j) o[j] = (__bf16)(ri * (float)rrow[k0 + j]);
  } else {
    int k0 = kk0 - 65536;
#pragma unroll
    for (int j = 0; j < 8; ++j) o[j] = rrow[k0 + j];
  }
  __bf16* dst = Afrag + (size_t)rest * 512 + (size_t)lane * 16 + half * 8;
  *(uint4*)dst = *(const uint4*)o;
}

// ======================================================================
// Per step: split-K bf16 WMMA GEMM with async global->LDS double buffer.
// Block = 8 waves = (mblk 0..3) x (2 nquads); block -> (npair, ks).
// Per K-tile the block stages 4 A-frags + 8 B-frags (12 KB) into LDS with
// global_load_async_to_lds_b128 (3 wave-instructions/wave, 512B each),
// overlapped with WMMA on the other buffer; consume via ds_load_b128.
// Chunk->region mapping is uniform per chunk index (0=A, 1..2=B), so the
// global pointers are hoisted and advanced by constant strides.
// ======================================================================
#define STAGE_BYTES 12288                    // 4*1024 (A) + 8*1024 (B)

#define ASYNC_B128(ldsoff, gaddr)                                       \
  asm volatile("global_load_async_to_lds_b128 %0, %1, off"              \
               :: "v"(ldsoff), "v"(gaddr) : "memory")

__global__ __launch_bounds__(256) void gemm_kernel(
    const __bf16* __restrict__ Afrag, const __bf16* __restrict__ Wc,
    float* __restrict__ part) {
  __shared__ __attribute__((aligned(32))) unsigned char smem[2 * STAGE_BYTES];

  const int tx    = threadIdx.x;
  const int lane  = tx & 31;
  const int wave  = tx >> 5;
  const int npair = blockIdx.x & 1;                  // columns npair*128..+127
  const int ks    = blockIdx.x >> 1;                 // 0..63
  const int mblk  = wave & 3;
  const int wq    = wave >> 2;                       // 0..1
  const int nquad = npair * 2 + wq;                  // 0..3 (64 cols)

  int kt0 = ks * 33;
  int kt1 = kt0 + 33; if (kt1 > KTILES) kt1 = KTILES;
  const int nkt = kt1 - kt0;

  v8f c0 = {}, c1 = {}, c2 = {}, c3 = {};

  const unsigned smem_base = (unsigned)(size_t)&smem[0];
  const unsigned char* aptr = &smem[0] + mblk * 1024 + lane * 32;
  const unsigned char* bptr = &smem[0] + 4096 + wq * 4096 + lane * 32;

  if (nkt > 0) {
    // Hoisted chunk pointers: chunk0 = A frag (stride 1KB/kt),
    // chunks 1,2 = B frags (stride 16KB/kt).
    const int sub = tx >> 6;                         // 0..3 within region
    const int byt = (tx & 63) * 16;                  // bytes within frag
    unsigned long long g0 = (unsigned long long)(const char*)Afrag
        + ((size_t)(sub * KTILES + kt0) << 10) + byt;
    unsigned long long g1 = (unsigned long long)(const char*)Wc
        + ((size_t)(kt0 * 16 + npair * 8 + sub) << 10) + byt;
    unsigned long long g2 = g1 + 4096;               // B frags l=4..7
    const unsigned l0 = (unsigned)(tx * 16);
    const unsigned l1 = l0 + 4096;
    const unsigned l2 = l0 + 8192;

    // prologue fill of buffer 0
    {
      unsigned base = smem_base;
      ASYNC_B128(base + l0, g0);
      ASYNC_B128(base + l1, g1);
      ASYNC_B128(base + l2, g2);
      g0 += 1024; g1 += 16384; g2 += 16384;
    }

    for (int i = 0; i < nkt; ++i) {
      int cur = i & 1;
      if (i + 1 < nkt) {
        unsigned base = smem_base + ((i + 1) & 1) * STAGE_BYTES;
        ASYNC_B128(base + l0, g0);
        ASYNC_B128(base + l1, g1);
        ASYNC_B128(base + l2, g2);
        g0 += 1024; g1 += 16384; g2 += 16384;
        asm volatile("s_wait_asynccnt 0x3" ::: "memory");
      } else {
        asm volatile("s_wait_asynccnt 0x0" ::: "memory");
      }
      __syncthreads();                               // stage[cur] visible

      const unsigned char* ab = aptr + cur * STAGE_BYTES;
      const unsigned char* bb = bptr + cur * STAGE_BYTES;
      v16bf a  = *(const v16bf*)ab;
      v16bf b0 = *(const v16bf*)(bb);
      v16bf b1 = *(const v16bf*)(bb + 1024);
      v16bf b2 = *(const v16bf*)(bb + 2048);
      v16bf b3 = *(const v16bf*)(bb + 3072);
      c0 = __builtin_amdgcn_wmma_f32_16x16x32_bf16(false, a, false, b0, (short)0, c0, false, false);
      c1 = __builtin_amdgcn_wmma_f32_16x16x32_bf16(false, a, false, b1, (short)0, c1, false, false);
      c2 = __builtin_amdgcn_wmma_f32_16x16x32_bf16(false, a, false, b2, (short)0, c2, false, false);
      c3 = __builtin_amdgcn_wmma_f32_16x16x32_bf16(false, a, false, b3, (short)0, c3, false, false);

      __syncthreads();                               // before overwriting stage[cur]
    }
  }

  // C/D layout: VGPR v -> M = v + 8*(lane>=16), N = lane&15
  float* pbase = part + (size_t)ks * (BB * HIDN)
                      + (size_t)(mblk * 16) * HIDN + nquad * 64;
  int M0 = (lane < 16) ? 0 : 8;
  int N  = lane & 15;
#pragma unroll
  for (int v = 0; v < 8; ++v) {
    float* row = pbase + (size_t)(M0 + v) * HIDN + N;
    row[0]  = c0[v];
    row[16] = c1[v];
    row[32] = c2[v];
    row[48] = c3[v];
  }
}

// ======================================================================
// Epilogue: out[b,t,:] = tanh(traj[b,t,:]) @ w_out_w.T + w_out_b
// ======================================================================
__global__ __launch_bounds__(256) void output_kernel(
    const float* __restrict__ traj, const float* __restrict__ w_out_w,
    const float* __restrict__ w_out_b, float* __restrict__ out) {
  __shared__ float sh[HIDN];
  int bt = blockIdx.x;                               // b*T + t
  sh[threadIdx.x] = tanhf(traj[(size_t)bt * HIDN + threadIdx.x]);
  __syncthreads();
  if (threadIdx.x < OUTN) {
    int o = threadIdx.x;
    const float* wrow = w_out_w + (size_t)o * HIDN;
    float s = w_out_b[o];
#pragma unroll 8
    for (int h = 0; h < HIDN; ++h) s += sh[h] * wrow[h];
    out[(size_t)bt * OUTN + o] = s;
  }
}

// ======================================================================
extern "C" void kernel_launch(void* const* d_in, const int* in_sizes, int n_in,
                              void* d_out, int out_size, void* d_ws, size_t ws_size,
                              hipStream_t stream) {
  const float* u         = (const float*)d_in[0];
  const float* x0        = (const float*)d_in[1];
  const float* noise     = (const float*)d_in[2];
  const float* w_hh_rnn  = (const float*)d_in[3];
  const float* w_hh_tbrnn= (const float*)d_in[4];
  const float* w_in_w    = (const float*)d_in[5];
  const float* w_in_b    = (const float*)d_in[6];
  const float* w_out_w   = (const float*)d_in[7];
  const float* w_out_b   = (const float*)d_in[8];

  float* out    = (float*)d_out;                     // [B,T,OUT]
  float* xfinal = out + (size_t)BB * TT * OUTN;      // [B,HID]
  float* traj   = xfinal + (size_t)BB * HIDN;        // [B,T,HID]

  char* ws = (char*)d_ws;
  float*  inputI = (float*)(ws + WS_INPUTI);
  __bf16* Wc     = (__bf16*)(ws + WS_WC);
  __bf16* Afrag  = (__bf16*)(ws + WS_AFRAG);
  __bf16* rbf    = (__bf16*)(ws + WS_RBF);
  float*  part   = (float*)(ws + WS_PART);

  convert_weights_kernel<<<(KTILES * 16 * 32) / 256, 256, 0, stream>>>(
      w_hh_tbrnn, w_hh_rnn, Wc);
  input_gemm_kernel<<<(TT * BB * HIDN) / 256, 256, 0, stream>>>(
      u, w_in_w, w_in_b, inputI);

  for (int t = 0; t <= TT; ++t) {
    update_kernel<<<(BB * HIDN) / 256, 256, 0, stream>>>(
        t, x0, noise, inputI, part, traj, xfinal, rbf);
    if (t < TT) {
      afrag_kernel<<<(4 * KTILES * 64) / 256, 256, 0, stream>>>(rbf, Afrag);
      gemm_kernel<<<128, 256, 0, stream>>>(Afrag, Wc, part);
    }
  }

  output_kernel<<<BB * TT, 256, 0, stream>>>(traj, w_out_w, w_out_b, out);
}
